// Model2_11596411699486
// MI455X (gfx1250) — matrix operand order repaired and checked by reading, not compiled, not run on hardware
//
#include <hip/hip_runtime.h>
#include <hip/hip_bf16.h>

// ---------- types ----------
typedef __attribute__((ext_vector_type(16))) __bf16        v16bf;
typedef __attribute__((ext_vector_type(8)))  float         v8f;
typedef __attribute__((ext_vector_type(4)))  unsigned int  v4u;
typedef __attribute__((ext_vector_type(8)))  unsigned int  v8u;
typedef unsigned long long ull;

__device__ __forceinline__ unsigned short f2bf(float f) {
    unsigned u = __builtin_bit_cast(unsigned, f);
    unsigned r = u + 0x7FFFu + ((u >> 16) & 1u);
    return (unsigned short)(r >> 16);
}

// 8-byte load from a 2-byte-aligned pointer; clang emits global_load_b64 when
// unaligned-access-mode is enabled, otherwise splits — correct either way.
__device__ __forceinline__ ull load8_a2(const unsigned short* p) {
    ull v;
    __builtin_memcpy(&v, p, 8);
    return v;
}

// ---------- fp32 -> bf16 elementwise ----------
__global__ __launch_bounds__(256) void cvt_bf16(const float* __restrict__ in,
                                                unsigned short* __restrict__ out, int n) {
    int t = blockIdx.x * 256 + threadIdx.x;
    if (t < n) out[t] = f2bf(in[t]);
}

// ---------- pack weights W[N][KT] (fp32) into WMMA B-fragment order ----------
// P flat layout: [ntile][kstep][lane(32)][e(16)]  (bf16)
// lane L holds column n = ntile*16 + (L&15); element e = B[k][n] with
// k = kstep*32 + (L<16 ? 0 : 8) + (e&7) + (e>=8 ? 16 : 0)   (matches v_wmma 16x16x32 B lane layout)
__global__ __launch_bounds__(256) void pack_w(const float* __restrict__ W,
                                              unsigned short* __restrict__ P,
                                              int N, int KT, int ksteps, int total) {
    int t = blockIdx.x * 256 + threadIdx.x;
    if (t >= total) return;
    int e     = t & 15;
    int lane  = (t >> 4) & 31;
    int blk   = t >> 9;                 // ntile*ksteps + s
    int ntile = blk / ksteps;
    int s     = blk - ntile * ksteps;
    int n     = ntile * 16 + (lane & 15);
    int k     = s * 32 + ((lane < 16) ? 0 : 8) + (e & 7) + ((e >> 3) << 4);
    float v   = (n < N && k < KT) ? W[(size_t)n * KT + k] : 0.f;
    P[t] = f2bf(v);
}

// ---------- implicit-GEMM 4D conv via WMMA bf16 ----------
// in : bf16 [256, CIN, DI^4]   out: bf16 [256, COUT, O^4] (ReLU(conv+bias))
// one wave = one 16(M) x 16(N) tile;  M = 256*O^4,  K = CIN*KS^4 in steps of 32
template <int CIN, int COUT, int DI, int KS>
__global__ __launch_bounds__(256) void conv_wmma(const unsigned short* __restrict__ xin,
                                                 const unsigned short* __restrict__ pB,
                                                 const float* __restrict__ bias,
                                                 unsigned short* __restrict__ out) {
    constexpr int O   = DI - KS + 1;
    constexpr int O2  = O * O, O3 = O2 * O, O4 = O3 * O;
    constexpr int D2  = DI * DI, D3 = D2 * DI, D4 = D3 * DI;
    constexpr int KS2 = KS * KS, KS3 = KS2 * KS, KS4 = KS3 * KS;
    constexpr int KT     = CIN * KS4;
    constexpr int KSTEPS = (KT + 31) / 32;
    constexpr int M      = 256 * O4;
    constexpr bool KMASK = (KT % 32) != 0;

    __shared__ __align__(16) unsigned short lds_a[8][16][32];

    const int lane = threadIdx.x & 31;
    const int w    = threadIdx.x >> 5;
    const int tile = blockIdx.x * 8 + w;
    if (tile * 16 >= M) return;
    const int m0 = tile * 16;

    // wave-uniform base address for each of the 16 M-rows of this tile
    unsigned myabase;
    {
        int m  = m0 + (lane & 15);
        int b  = m / O4;  int r = m - b * O4;
        int o1 = r / O3;  r -= o1 * O3;
        int o2 = r / O2;  r -= o2 * O2;
        int o3 = r / O;   int o4 = r - o3 * O;
        myabase = (unsigned)(b * CIN) * D4 + o1 * D3 + o2 * D2 + o3 * DI + o4;
    }
    unsigned abase[16];
#pragma unroll
    for (int e = 0; e < 16; ++e) abase[e] = __shfl(myabase, e, 32);

    unsigned short* aslot = &lds_a[w][0][0];
    v8f acc = {};

    for (int s = 0; s < KSTEPS; ++s) {
        if constexpr (KS == 4) {
            // k%256 decomposes by shifts; a 16-aligned k-base means j3 = run
            // index, j4 = 0, so lane L gathers row L>>1, 16 contiguous k-values
            // = 4 runs of 4 bf16 that are contiguous in memory (j4 innermost).
            const int row    = lane >> 1;
            const int base16 = (lane & 1) * 16;
            const int k0     = s * 32 + base16;
            const int cin    = k0 >> 8;
            const int r      = k0 & 255;
            const int j1     = r >> 6;
            const int j2     = (r >> 4) & 3;
            const unsigned koff = (unsigned)cin * D4 + j1 * D3 + j2 * D2; // j3=i, j4=0
            const unsigned short* src = xin + (size_t)abase[row] + koff;
            ull v0 = load8_a2(src);
            ull v1 = load8_a2(src + DI);
            ull v2 = load8_a2(src + 2 * DI);
            ull v3 = load8_a2(src + 3 * DI);
            v4u q0, q1;
            q0[0] = (unsigned)v0; q0[1] = (unsigned)(v0 >> 32);
            q0[2] = (unsigned)v1; q0[3] = (unsigned)(v1 >> 32);
            q1[0] = (unsigned)v2; q1[1] = (unsigned)(v2 >> 32);
            q1[2] = (unsigned)v3; q1[3] = (unsigned)(v3 >> 32);
            *(v4u*)(&lds_a[w][row][base16])     = q0;
            *(v4u*)(&lds_a[w][row][base16 + 8]) = q1;
        } else {
            // generic path (KS==3): lane = K column, scalar gather over 16 rows
            const int k = s * 32 + lane;
            const bool kvalid = !KMASK || (k < KT);
            unsigned koff;
            {
                int cin = k / KS4;  int r = k - cin * KS4;
                int j1  = r / KS3;  r -= j1 * KS3;
                int j2  = r / KS2;  r -= j2 * KS2;
                int j3  = r / KS;   int j4 = r - j3 * KS;
                koff = (unsigned)cin * D4 + j1 * D3 + j2 * D2 + j3 * DI + j4;
            }
#pragma unroll
            for (int e = 0; e < 16; ++e) {
                unsigned short v = kvalid ? xin[(size_t)abase[e] + koff] : (unsigned short)0;
                aslot[e * 32 + lane] = v;
            }
        }
        __builtin_amdgcn_wave_barrier();   // LDS is in-order per wave; pin instr order

        // A fragment per ISA lane layout (two ds_load_b128)
        const int  mrow = lane & 15;
        const int  kb   = (lane < 16) ? 0 : 8;
        const v4u* ap   = (const v4u*)(&lds_a[w][mrow][kb]);
        v4u alo = ap[0];
        v4u ahi = ap[2];        // +16 halves
        v8u au;
        au[0]=alo[0]; au[1]=alo[1]; au[2]=alo[2]; au[3]=alo[3];
        au[4]=ahi[0]; au[5]=ahi[1]; au[6]=ahi[2]; au[7]=ahi[3];
        v16bf afrag = __builtin_bit_cast(v16bf, au);

        // B fragment: contiguous 32B per lane from packed weights (L2-resident)
        const v4u* bp = (const v4u*)(pB + ((size_t)s * 512 + lane * 16));
        v4u blo = bp[0];
        v4u bhi = bp[1];
        v8u bu;
        bu[0]=blo[0]; bu[1]=blo[1]; bu[2]=blo[2]; bu[3]=blo[3];
        bu[4]=bhi[0]; bu[5]=bhi[1]; bu[6]=bhi[2]; bu[7]=bhi[3];
        v16bf bfrag = __builtin_bit_cast(v16bf, bu);

        acc = __builtin_amdgcn_wmma_f32_16x16x32_bf16(false, afrag, false, bfrag,
                                                      (short)0, acc, false, false);
        __builtin_amdgcn_wave_barrier();   // WAR: next gather overwrites LDS tile
    }

    // store D: lane holds column n=lane&15, rows m0 + r + (lane<16?0:8)
    const int n = lane & 15;
    if (n < COUT) {
        const float bv = bias[n];
#pragma unroll
        for (int r = 0; r < 8; ++r) {
            int m    = m0 + r + ((lane < 16) ? 0 : 8);
            int b    = m / O4;
            int spat = m - b * O4;
            size_t addr = (size_t)b * (COUT * O4) + (size_t)n * O4 + spat;
            float v = acc[r] + bv;
            v = v > 0.f ? v : 0.f;
            out[addr] = f2bf(v);
        }
    }
}

// ---------- dense1: [256 x 3840] @ [3840 x 100] + bias, ReLU -> fp32 [256][100]
__global__ __launch_bounds__(256) void dense1_wmma(const unsigned short* __restrict__ A,
                                                   const unsigned short* __restrict__ pB,
                                                   const float* __restrict__ bias,
                                                   float* __restrict__ out) {
    const int lane = threadIdx.x & 31;
    const int w    = threadIdx.x >> 5;
    const int gw   = blockIdx.x * 8 + w;      // 0..111 : 16 mtiles x 7 ntiles
    if (gw >= 112) return;
    const int mtile = gw / 7;
    const int ntile = gw - mtile * 7;

    const int m  = mtile * 16 + (lane & 15);
    const int kb = (lane < 16) ? 0 : 8;
    v8f acc = {};
    for (int s = 0; s < 120; ++s) {
        const v4u* ap = (const v4u*)(A + ((size_t)m * 3840 + s * 32 + kb));
        v4u alo = ap[0], ahi = ap[2];
        v8u au;
        au[0]=alo[0]; au[1]=alo[1]; au[2]=alo[2]; au[3]=alo[3];
        au[4]=ahi[0]; au[5]=ahi[1]; au[6]=ahi[2]; au[7]=ahi[3];
        v16bf afrag = __builtin_bit_cast(v16bf, au);

        const v4u* bp = (const v4u*)(pB + ((size_t)ntile * 120 * 512 + (size_t)s * 512 + lane * 16));
        v4u blo = bp[0], bhi = bp[1];
        v8u bu;
        bu[0]=blo[0]; bu[1]=blo[1]; bu[2]=blo[2]; bu[3]=blo[3];
        bu[4]=bhi[0]; bu[5]=bhi[1]; bu[6]=bhi[2]; bu[7]=bhi[3];
        v16bf bfrag = __builtin_bit_cast(v16bf, bu);

        acc = __builtin_amdgcn_wmma_f32_16x16x32_bf16(false, afrag, false, bfrag,
                                                      (short)0, acc, false, false);
    }
    const int n = ntile * 16 + (lane & 15);
    if (n < 100) {
        const float bv = bias[n];
#pragma unroll
        for (int r = 0; r < 8; ++r) {
            int mm  = mtile * 16 + r + ((lane < 16) ? 0 : 8);
            float v = acc[r] + bv;
            v = v > 0.f ? v : 0.f;
            out[(size_t)mm * 100 + n] = v;
        }
    }
}

// ---------- dense2 + softmax ----------
__global__ __launch_bounds__(256) void head_kernel(const float* __restrict__ h6,
                                                   const float* __restrict__ W2,
                                                   const float* __restrict__ b2,
                                                   float* __restrict__ out) {
    int b = blockIdx.x * 256 + threadIdx.x;
    if (b >= 256) return;
    float l0 = b2[0], l1 = b2[1];
    for (int k = 0; k < 100; ++k) {
        float h = h6[b * 100 + k];
        l0 += h * W2[k];
        l1 += h * W2[100 + k];
    }
    float mx = l0 > l1 ? l0 : l1;
    float e0 = __expf(l0 - mx), e1 = __expf(l1 - mx);
    float s = e0 + e1;
    out[b * 2 + 0] = e0 / s;
    out[b * 2 + 1] = e1 / s;
}

// ---------- host launcher ----------
extern "C" void kernel_launch(void* const* d_in, const int* in_sizes, int n_in,
                              void* d_out, int out_size, void* d_ws, size_t ws_size,
                              hipStream_t stream) {
    const float* x   = (const float*)d_in[0];
    const float* w1  = (const float*)d_in[1];  const float* b1 = (const float*)d_in[2];
    const float* w2  = (const float*)d_in[3];  const float* b2 = (const float*)d_in[4];
    const float* w3  = (const float*)d_in[5];  const float* b3 = (const float*)d_in[6];
    const float* w4  = (const float*)d_in[7];  const float* b4 = (const float*)d_in[8];
    const float* w5  = (const float*)d_in[9];  const float* b5 = (const float*)d_in[10];
    const float* dW1 = (const float*)d_in[11]; const float* db1 = (const float*)d_in[12];
    const float* dW2 = (const float*)d_in[13]; const float* db2 = (const float*)d_in[14];
    float* out = (float*)d_out;

    // workspace layout (bytes): ping-pong bf16 activation regions + packs
    char* ws = (char*)d_ws;
    const size_t off_r0   = 0;                        // max(xb 53.7MB, h2 95.5MB, h4 10MB)
    const size_t off_r1   = 95551488;                 // max(h1 77.8MB, h3 40.3MB, h5 2MB)
    const size_t off_h6   = off_r1 + 77760000;        // fp32 [256][100]
    const size_t off_pack = off_h6 + 102400;
    unsigned short* r0 = (unsigned short*)(ws + off_r0);
    unsigned short* r1 = (unsigned short*)(ws + off_r1);
    float*          h6 = (float*)(ws + off_h6);
    unsigned short* pw1 = (unsigned short*)(ws + off_pack);            //   8*512
    unsigned short* pw2 = pw1 + 8   * 512;                             //  24*512
    unsigned short* pw3 = pw2 + 24  * 512;                             //  72*512
    unsigned short* pw4 = pw3 + 72  * 512;                             //  96*512
    unsigned short* pw5 = pw4 + 96  * 512;                             //  38*512
    unsigned short* pd1 = pw5 + 38  * 512;                             // 7*120*512

    // 1) input fp32 -> bf16
    {
        int n = 256 * 1 * 18 * 18 * 18 * 18;   // 26,873,856
        cvt_bf16<<<(n + 255) / 256, 256, 0, stream>>>(x, r0, n);
    }
    // 2) pack all weight matrices into WMMA B order
    auto pack = [&](const float* W, unsigned short* P, int N, int KT, int ntiles, int ksteps) {
        int total = ntiles * ksteps * 512;
        pack_w<<<(total + 255) / 256, 256, 0, stream>>>(W, P, N, KT, ksteps, total);
    };
    pack(w1,  pw1, 3,   256,  1, 8);
    pack(w2,  pw2, 9,   768,  1, 24);
    pack(w3,  pw3, 12,  2304, 1, 72);
    pack(w4,  pw4, 15,  3072, 1, 96);
    pack(w5,  pw5, 15,  1215, 1, 38);
    pack(dW1, pd1, 100, 3840, 7, 120);

    // 3) conv stack (one wave per 16x16 tile; 8 waves/block)
    conv_wmma<1,  3,  18, 4><<<101250, 256, 0, stream>>>(r0, pw1, b1, r1);  // -> [256,3,15^4]
    conv_wmma<3,  9,  15, 4><<<41472,  256, 0, stream>>>(r1, pw2, b2, r0);  // -> [256,9,12^4]
    conv_wmma<9,  12, 12, 4><<<13122,  256, 0, stream>>>(r0, pw3, b3, r1);  // -> [256,12,9^4]
    conv_wmma<12, 15, 9,  4><<<2592,   256, 0, stream>>>(r1, pw4, b4, r0);  // -> [256,15,6^4]
    conv_wmma<15, 15, 6,  3><<<512,    256, 0, stream>>>(r0, pw5, b5, r1);  // -> [256,15,4^4]=h5

    // 4) dense1 (WMMA) + head
    dense1_wmma<<<14, 256, 0, stream>>>(r1, pd1, db1, h6);
    head_kernel<<<1, 256, 0, stream>>>(h6, dW2, db2, out);
}
